// SelfAttention_77369540870183
// MI455X (gfx1250) — matrix operand order repaired and checked by reading, not compiled, not run on hardware
//
#include <hip/hip_runtime.h>

// ---------------------------------------------------------------------------
// Self-attention, B=8 S=2048 D=512, fp32 in/out.
// bf16 WMMA (v_wmma_f32_16x16x32_bf16) for all GEMM stages; flash-attention
// online softmax; TDM (tensor_load_to_lds) double-buffered K-tile staging.
// ---------------------------------------------------------------------------

typedef __attribute__((ext_vector_type(16))) __bf16 v16bf;
typedef __attribute__((ext_vector_type(8)))  float  v8f;
typedef __attribute__((ext_vector_type(4)))  unsigned int v4u;
typedef __attribute__((ext_vector_type(8)))  int v8i;
typedef __attribute__((ext_vector_type(4)))  int v4i;

#define DDIM 512
#define SDIM 2048
#define BDIM 8

#if defined(__has_builtin)
#  if __has_builtin(__builtin_amdgcn_tensor_load_to_lds) && \
      __has_builtin(__builtin_amdgcn_s_wait_tensorcnt)
#    define HAVE_TDM 1
#  endif
#endif
#ifndef HAVE_TDM
#  define HAVE_TDM 0
#endif

union BF16x16 { v16bf v; unsigned u[8]; };

static __device__ inline unsigned pack2(__bf16 lo, __bf16 hi) {
  union { __bf16 b[2]; unsigned u; } t;
  t.b[0] = lo; t.b[1] = hi;
  return t.u;
}

// A fragment (16x32 bf16, MxK), source row-major [row][k], leading dim ld.
// ISA 7.12.2: M = lane&15; VGPR v holds K = (v<4?0:16) + 8*(lane>>4) + (v&3)*2.
static __device__ inline v16bf load_a_frag(const __bf16* base, int ld) {
  int lane = threadIdx.x & 31;
  int r = lane & 15, h = lane >> 4;
  const __bf16* p = base + (size_t)r * ld;
  BF16x16 f;
#pragma unroll
  for (int v = 0; v < 8; ++v) {
    int kk = ((v < 4) ? 0 : 16) + h * 8 + (v & 3) * 2;
    f.u[v] = *(const unsigned*)(p + kk);
  }
  return f.v;
}

// B fragment (32x16 bf16, KxN) from an N-major source: src[n][k] contiguous in
// k, leading dim ld. Layout: lane = h*16+n; VGPR v holds K = (v<4?0:16)+8h+2(v&3) pair, col n.
static __device__ inline v16bf load_b_frag_nk(const __bf16* base, int ld) {
  int lane = threadIdx.x & 31;
  int n = lane & 15, h = lane >> 4;
  const __bf16* p = base + (size_t)n * ld + h * 8;
  BF16x16 f;
#pragma unroll
  for (int v = 0; v < 8; ++v)
    f.u[v] = *(const unsigned*)(p + ((v < 4) ? 0 : 16) + 2 * (v & 3));
  return f.v;
}

#if HAVE_TDM
// TDM: DMA a 32-row x 512-col bf16 tile (row stride 512 elems, contiguous)
// from global into LDS. D# per CDNA5 ISA §8.3/8.4. 6-arg builtin form
// (clang-23 / therock-10.0): (g0, g1, g2, g3, extra, cpol).
static __device__ inline void tdm_load_k_tile(const __bf16* gsrc, unsigned lds_off) {
  unsigned long long ga = (unsigned long long)(uintptr_t)gsrc;
  v4u g0;
  g0.x = 1u;                                             // count=1, user D#
  g0.y = lds_off;                                        // lds_addr (bytes)
  g0.z = (unsigned)(ga & 0xFFFFFFFFull);                 // global_addr[31:0]
  g0.w = ((unsigned)(ga >> 32) & 0x01FFFFFFu)            // global_addr[56:32]
         | 0x80000000u;                                  // type=2 ("image")
  v8i g1;
  g1[0] = (int)(1u << 16);                     // data_size=1 (2 bytes/elem)
  g1[1] = (int)(512u << 16);                   // tensor_dim0 = 512 (lo16)
  g1[2] = (int)(2048u << 16);                  // dim0 hi16=0 | tensor_dim1 lo16
  g1[3] = (int)(512u << 16);                   // dim1 hi16=0 | tile_dim0 = 512
  g1[4] = 32;                                  // tile_dim1 = 32 rows
  g1[5] = 512;                                 // tensor_dim0_stride = 512
  g1[6] = 0;
  g1[7] = 0;
  v4i z4 = {0, 0, 0, 0};                       // groups 2/3 unused (2D tensor)
  v8i z8 = {0, 0, 0, 0, 0, 0, 0, 0};
  __builtin_amdgcn_tensor_load_to_lds(g0, g1, z4, z4, z8, 0);
}
#endif

// ---------------------------------------------------------------------------
// Conversion kernels
// ---------------------------------------------------------------------------
__global__ void cvt_bf16(const float* __restrict__ s, __bf16* __restrict__ d, int n) {
  int i = blockIdx.x * blockDim.x + threadIdx.x;
  if (i < n) d[i] = (__bf16)s[i];
}

// W is [d][e] (512x512); store transposed bf16 [e][d] so GEMM B-frags read
// contiguous K pairs.
__global__ void cvt_bf16_T512(const float* __restrict__ s, __bf16* __restrict__ d) {
  int i = blockIdx.x * blockDim.x + threadIdx.x;  // 262144 threads
  int row = i >> 9, col = i & 511;                // row=d, col=e
  d[(size_t)col * DDIM + row] = (__bf16)s[i];
}

// ---------------------------------------------------------------------------
// Fused QKV projection: out = x @ W + b, bf16 storage.
// grid = (M/128, 512/64, 3); block = 256 (8 waves). Wave w -> 16x64 tile.
// z==0 -> Q (pre-scaled by 1/sqrt(D)), z==1 -> K, z==2 -> V transposed [B,D,S].
// ---------------------------------------------------------------------------
__global__ __launch_bounds__(256) void qkv_gemm(
    const __bf16* __restrict__ xb,
    const __bf16* __restrict__ WqT,
    const __bf16* __restrict__ WkT,
    const __bf16* __restrict__ WvT,
    const float* __restrict__ bq,
    const float* __restrict__ bk,
    const float* __restrict__ bv,
    __bf16* __restrict__ Qb,
    __bf16* __restrict__ Kb,
    __bf16* __restrict__ Vt)
{
  int z = blockIdx.z;
  const __bf16* WT  = (z == 0) ? WqT : ((z == 1) ? WkT : WvT);
  const float* bias = (z == 0) ? bq  : ((z == 1) ? bk  : bv);
  float scale = (z == 0) ? 0.04419417382415922f : 1.0f;  // 1/sqrt(512)

  int w = threadIdx.x >> 5, lane = threadIdx.x & 31;
  int m0 = blockIdx.x * 128 + w * 16;
  int n0 = blockIdx.y * 64;
  int r = lane & 15, h = lane >> 4;

  v8f acc[4] = {};
  for (int k0 = 0; k0 < DDIM; k0 += 32) {
    // L2/L0 prefetch of the next A strip (activation stream).
    if (k0 + 64 < DDIM)
      __builtin_prefetch(xb + (size_t)(m0 + r) * DDIM + k0 + 64, 0, 3);
    v16bf a = load_a_frag(xb + (size_t)m0 * DDIM + k0, DDIM);
#pragma unroll
    for (int t = 0; t < 4; ++t) {
      v16bf b = load_b_frag_nk(WT + (size_t)(n0 + t * 16) * DDIM + k0, DDIM);
      acc[t] = __builtin_amdgcn_wmma_f32_16x16x32_bf16(
          false, a, false, b, (short)0, acc[t], false, false);
    }
  }

  int n = lane & 15;
#pragma unroll
  for (int t = 0; t < 4; ++t) {
    int col = n0 + t * 16 + n;
    float bb = bias[col];
#pragma unroll
    for (int v = 0; v < 8; ++v) {
      int row = m0 + v + 8 * h;
      __bf16 o = (__bf16)((acc[t][v] + bb) * scale);
      if (z == 2) {
        int b_ = row >> 11;
        int s_ = row & 2047;
        Vt[((size_t)b_ * DDIM + col) * SDIM + s_] = o;
      } else if (z == 0) {
        Qb[(size_t)row * DDIM + col] = o;
      } else {
        Kb[(size_t)row * DDIM + col] = o;
      }
    }
  }
}

// ---------------------------------------------------------------------------
// Flash attention. grid = (S/16, B); block = 256 (8 waves).
// Block: 16 query rows; wave w owns output columns [64w, 64w+64).
// Per 32-key tile: TDM-staged K tile (double-buffered), per-wave partial
// scores over its d-slice, ds_add_f32 cross-wave reduction, wave-0 online
// softmax (shfl_xor halves), accumulator rescale, P(bf16) x V WMMAs.
// ---------------------------------------------------------------------------
__global__ __launch_bounds__(256) void flash_attn(
    const __bf16* __restrict__ Qb,   // [B*S, 512], pre-scaled by 1/sqrt(D)
    const __bf16* __restrict__ Kb,   // [B*S, 512]
    const __bf16* __restrict__ Vt,   // [B][512][S]
    float* __restrict__ Out)         // [B*S, 512]
{
#if HAVE_TDM
  __shared__ __bf16 ldsK[2][32 * DDIM];          // 2 x 32KB K-tile buffers
#endif
  __shared__ float sc[16][32];                   // fp32 score tile
  __shared__ unsigned pb[16][16];                // packed bf16 prob pairs
  __shared__ float row_m[16], row_l[16], row_a[16];

  int bz = blockIdx.y;
  int q0 = blockIdx.x * 16;
  int w = threadIdx.x >> 5, lane = threadIdx.x & 31;
  int d0 = w * 64;
  int n = lane & 15, h = lane >> 4;

  const __bf16* Qbase = Qb + ((size_t)bz * SDIM + q0) * DDIM;
  const __bf16* Kbase = Kb + (size_t)bz * SDIM * DDIM;
  const __bf16* Vbase = Vt + (size_t)bz * DDIM * SDIM;

  // Q fragments for this wave's 64-wide d slice (loop invariant, 16 VGPRs).
  v16bf aq0 = load_a_frag(Qbase + d0,      DDIM);
  v16bf aq1 = load_a_frag(Qbase + d0 + 32, DDIM);

  // Init score buffer + running softmax state.
  ((float*)sc)[threadIdx.x]       = 0.0f;
  ((float*)sc)[threadIdx.x + 256] = 0.0f;
  if (threadIdx.x < 16) { row_m[threadIdx.x] = -3.0e38f; row_l[threadIdx.x] = 0.0f; }

#if HAVE_TDM
  unsigned lds_k_base = (unsigned)(uintptr_t)(&ldsK[0][0]);
  if (w == 0) tdm_load_k_tile(Kbase, lds_k_base);          // tile 0 -> buf 0
#endif

  v8f acc[4] = {};  // output accumulators: 4 tiles of 16x16 over cols d0..d0+63

  const int NT = SDIM / 32;
  for (int it = 0; it < NT; ++it) {
    int k0 = it * 32;
#if HAVE_TDM
    int buf = it & 1;
    if (w == 0) {
      if (it + 1 < NT) {
        tdm_load_k_tile(Kbase + (size_t)(k0 + 32) * DDIM,
                        lds_k_base + (unsigned)((buf ^ 1) * 32 * DDIM * 2));
        __builtin_amdgcn_s_wait_tensorcnt(1);   // tile `it` complete
      } else {
        __builtin_amdgcn_s_wait_tensorcnt(0);
      }
    }
#endif
    __syncthreads();   // K tile visible; sc zeroed; prev-iter pb reads done

#if HAVE_TDM
    const __bf16* Ktile = &ldsK[it & 1][0];                // [key_rel][d]
#else
    const __bf16* Ktile = Kbase + (size_t)k0 * DDIM;
#endif

    // Partial scores over this wave's d slice: 2 key sub-tiles x 2 k-steps.
    v8f ps[2] = {};
#pragma unroll
    for (int t = 0; t < 2; ++t) {
      v16bf b0 = load_b_frag_nk(Ktile + (size_t)(t * 16) * DDIM + d0, DDIM);
      ps[t] = __builtin_amdgcn_wmma_f32_16x16x32_bf16(
          false, aq0, false, b0, (short)0, ps[t], false, false);
      v16bf b1 = load_b_frag_nk(Ktile + (size_t)(t * 16) * DDIM + d0 + 32, DDIM);
      ps[t] = __builtin_amdgcn_wmma_f32_16x16x32_bf16(
          false, aq1, false, b1, (short)0, ps[t], false, false);
    }

    // Cross-wave reduction of partial scores into LDS.
#pragma unroll
    for (int t = 0; t < 2; ++t)
#pragma unroll
      for (int v = 0; v < 8; ++v)
        atomicAdd(&sc[v + 8 * h][t * 16 + n], ps[t][v]);
    __syncthreads();

    // Online softmax: wave 0, lane = (row r, half hh) covers 16 columns.
    if (w == 0) {
      int r = lane & 15, hh = lane >> 4;
      float mo = row_m[r];
      float sv[16];
      float mt = -3.0e38f;
#pragma unroll
      for (int j = 0; j < 16; ++j) {
        sv[j] = sc[r][hh * 16 + j];
        mt = fmaxf(mt, sv[j]);
      }
      mt = fmaxf(mt, __shfl_xor(mt, 16));
      float mnew = fmaxf(mo, mt);
      float lsum = 0.0f;
#pragma unroll
      for (int j = 0; j < 16; j += 2) {
        float p0 = __expf(sv[j]     - mnew);
        float p1 = __expf(sv[j + 1] - mnew);
        lsum += p0 + p1;
        pb[r][hh * 8 + (j >> 1)] = pack2((__bf16)p0, (__bf16)p1);
      }
      lsum += __shfl_xor(lsum, 16);
      // Re-zero sc for the next tile's atomic reduction.
#pragma unroll
      for (int j = 0; j < 16; ++j) sc[r][hh * 16 + j] = 0.0f;
      if (hh == 0) {
        float alpha = __expf(mo - mnew);
        row_m[r] = mnew;
        row_l[r] = row_l[r] * alpha + lsum;
        row_a[r] = alpha;
      }
    }
    __syncthreads();

    // Rescale running output accumulators by alpha(row).
    float al[8];
#pragma unroll
    for (int v = 0; v < 8; ++v) al[v] = row_a[v + 8 * h];
#pragma unroll
    for (int t = 0; t < 4; ++t)
#pragma unroll
      for (int v = 0; v < 8; ++v) acc[t][v] *= al[v];

    // P as a bf16 A-fragment (16x32) from packed pairs.
    BF16x16 pf;
    {
      int r = lane & 15;
#pragma unroll
      for (int v = 0; v < 8; ++v) {
        int kk2 = ((v < 4) ? 0 : 8) + h * 4 + (v & 3);   // pair index kk/2
        pf.u[v] = pb[r][kk2];
      }
    }

    // acc += P @ V_tile (V transposed: contiguous key pairs).
#pragma unroll
    for (int t = 0; t < 4; ++t) {
      v16bf bvf = load_b_frag_nk(Vbase + (size_t)(d0 + t * 16) * SDIM + k0, SDIM);
      acc[t] = __builtin_amdgcn_wmma_f32_16x16x32_bf16(
          false, pf.v, false, bvf, (short)0, acc[t], false, false);
    }
  }
  __syncthreads();

  // Normalize and store fp32 output.
  float li[8];
#pragma unroll
  for (int v = 0; v < 8; ++v) li[v] = 1.0f / row_l[v + 8 * h];
  float* Obase = Out + ((size_t)bz * SDIM + q0) * DDIM;
#pragma unroll
  for (int t = 0; t < 4; ++t)
#pragma unroll
    for (int v = 0; v < 8; ++v)
      Obase[(size_t)(v + 8 * h) * DDIM + d0 + t * 16 + n] = acc[t][v] * li[v];
}

// ---------------------------------------------------------------------------
extern "C" void kernel_launch(void* const* d_in, const int* in_sizes, int n_in,
                              void* d_out, int out_size, void* d_ws, size_t ws_size,
                              hipStream_t stream) {
  (void)in_sizes; (void)n_in; (void)out_size; (void)ws_size;
  const int M = BDIM * SDIM;                 // 16384
  const float* x  = (const float*)d_in[0];
  const float* Wq = (const float*)d_in[1];
  const float* bq = (const float*)d_in[2];
  const float* Wk = (const float*)d_in[3];
  const float* bk = (const float*)d_in[4];
  const float* Wv = (const float*)d_in[5];
  const float* bv = (const float*)d_in[6];

  char* ws = (char*)d_ws;
  size_t o = 0;
  __bf16* xb  = (__bf16*)(ws + o); o += (size_t)M * DDIM * 2;        // 16 MB
  __bf16* WqT = (__bf16*)(ws + o); o += (size_t)DDIM * DDIM * 2;     // 512 KB
  __bf16* WkT = (__bf16*)(ws + o); o += (size_t)DDIM * DDIM * 2;
  __bf16* WvT = (__bf16*)(ws + o); o += (size_t)DDIM * DDIM * 2;
  __bf16* Qb  = (__bf16*)(ws + o); o += (size_t)M * DDIM * 2;        // 16 MB
  __bf16* Kb  = (__bf16*)(ws + o); o += (size_t)M * DDIM * 2;        // 16 MB
  __bf16* Vt  = (__bf16*)(ws + o); o += (size_t)M * DDIM * 2;        // 16 MB

  int nx = M * DDIM;
  cvt_bf16<<<(nx + 255) / 256, 256, 0, stream>>>(x, xb, nx);
  int nw = DDIM * DDIM;
  cvt_bf16_T512<<<(nw + 255) / 256, 256, 0, stream>>>(Wq, WqT);
  cvt_bf16_T512<<<(nw + 255) / 256, 256, 0, stream>>>(Wk, WkT);
  cvt_bf16_T512<<<(nw + 255) / 256, 256, 0, stream>>>(Wv, WvT);

  dim3 g1(M / 128, DDIM / 64, 3);
  qkv_gemm<<<g1, 256, 0, stream>>>(xb, WqT, WkT, WvT, bq, bk, bv, Qb, Kb, Vt);

  dim3 g2(SDIM / 16, BDIM);
  flash_attn<<<g2, 256, 0, stream>>>(Qb, Kb, Vt, (float*)d_out);
}
